// ManualLSTMDecryptor_35923106464033
// MI455X (gfx1250) — compile-verified, matrix operand order checked by reading, not run on hardware
//
#include <hip/hip_runtime.h>
#include <hip/hip_bf16.h>

typedef __attribute__((ext_vector_type(16))) __bf16 v16bf;
typedef __attribute__((ext_vector_type(8)))  float  v8f;

#define B_    128
#define T_    1024
#define H_    512
#define V_    256
#define KZ    1024   /* E + H */
#define NG    2048   /* 4H    */
#define NBLK  32
#define NTHR  256

/* ---------------- workspace layout (bytes) ---------------- */
static constexpr size_t OFF_WBF   = 0;
static constexpr size_t SZ_WBF    = (size_t)2 * NG * KZ * 2;       // 8 MB bf16 LSTM weights
static constexpr size_t OFF_PWBF  = OFF_WBF + SZ_WBF;
static constexpr size_t SZ_PWBF   = (size_t)V_ * H_ * 2;           // proj weights bf16
static constexpr size_t OFF_EMBBF = OFF_PWBF + SZ_PWBF;
static constexpr size_t SZ_EMBBF  = (size_t)256 * 512 * 2;         // emb table bf16
static constexpr size_t SZ_ACT    = (size_t)B_ * H_ * 2;           // one activation buffer
static constexpr size_t OFF_XE    = OFF_EMBBF + SZ_EMBBF;
static constexpr size_t OFF_H0A   = OFF_XE  + SZ_ACT;
static constexpr size_t OFF_H0B   = OFF_H0A + SZ_ACT;
static constexpr size_t OFF_H1A   = OFF_H0B + SZ_ACT;
static constexpr size_t OFF_H1B   = OFF_H1A + SZ_ACT;
static constexpr size_t OFF_LNA   = OFF_H1B + SZ_ACT;
static constexpr size_t SZ_C      = (size_t)B_ * H_ * 4;
static constexpr size_t OFF_C0    = OFF_LNA + SZ_ACT;
static constexpr size_t OFF_C1    = OFF_C0 + SZ_C;
static constexpr size_t OFF_BAR   = OFF_C1 + SZ_C;                 // 256 B barrier block

union FragA { v16bf v; uint4 q[2]; };

__device__ inline float sigm(float x) {
  return __builtin_amdgcn_rcpf(1.0f + __expf(-x));
}
__device__ inline float tanh_f(float x) {
  float xc = fminf(fmaxf(x, -15.0f), 15.0f);
  float e  = __expf(2.0f * xc);
  return (e - 1.0f) * __builtin_amdgcn_rcpf(e + 1.0f);
}

/* ------------- device-wide sense barrier (persistent grid) ------------- */
__device__ inline void gbar(char* ws) {
  volatile unsigned* gen = (volatile unsigned*)(ws + OFF_BAR);
  unsigned* cnt = (unsigned*)(ws + OFF_BAR + 64);
  __syncthreads();
  if (threadIdx.x == 0) {
    __threadfence();
    unsigned g = *gen;
    if (atomicAdd(cnt, 1u) == NBLK - 1u) {
      atomicExch(cnt, 0u);
      __threadfence();
      atomicAdd((unsigned*)gen, 1u);
    } else {
      while (*gen == g) { __builtin_amdgcn_s_sleep(1); }
    }
  }
  __syncthreads();
}

/* ------------- fused gate-GEMM + LSTM cell: one wave = one 16x16 h-tile,
   computes its i/f/o/g gate tiles (n, n+512, n+1024, n+1536) sharing A ------------- */
__device__ inline void gemm_cell(int gw, int lane,
                                 const __bf16* __restrict__ inp,
                                 const __bf16* __restrict__ hprev,
                                 __bf16* __restrict__ hnext,
                                 float* __restrict__ c,
                                 const __bf16* __restrict__ Wl,
                                 const float* __restrict__ bias)
{
  const int half  = lane >> 4, r = lane & 15;
  const int mt    = gw >> 5, ntg = gw & 31;           // 8 x 32 tile groups
  const int mBase = mt * 16, nBase = ntg * 16;
  const int col   = nBase + r;
  const int kl    = half * 8;

  const float b0 = bias[col], b1 = bias[512 + col], b2 = bias[1024 + col], b3 = bias[1536 + col];
  v8f ai = {b0, b0, b0, b0, b0, b0, b0, b0};
  v8f af = {b1, b1, b1, b1, b1, b1, b1, b1};
  v8f ao = {b2, b2, b2, b2, b2, b2, b2, b2};
  v8f ag = {b3, b3, b3, b3, b3, b3, b3, b3};

  #pragma unroll 4
  for (int kc = 0; kc < 32; ++kc) {
    // A fragment: z = concat(inp, hprev); each 32-wide K chunk sits fully in one half
    const __bf16* src = (kc < 16) ? inp : hprev;
    const int k0 = ((kc & 15) << 5) + kl;
    FragA A;
    { const uint4* p = (const uint4*)(src + (size_t)(mBase + r) * H_ + k0);
      A.q[0] = p[0]; A.q[1] = p[2]; }                 // K {0..7} and {16..23} (per half)

    const int kw = (kc << 5) + kl;                    // full 1024-wide K in W rows
    FragA Bq[4];
    #pragma unroll
    for (int g = 0; g < 4; ++g) {
      const uint4* p = (const uint4*)(Wl + (size_t)(g * 512 + nBase + r) * KZ + kw);
      Bq[g].q[0] = p[0]; Bq[g].q[1] = p[2];
    }
    if (kc + 1 < 32)
      __builtin_prefetch(Wl + (size_t)(nBase + r) * KZ + ((kc + 1) << 5), 0, 3);

    ai = __builtin_amdgcn_wmma_f32_16x16x32_bf16(false, A.v, false, Bq[0].v, (short)0, ai, false, false);
    af = __builtin_amdgcn_wmma_f32_16x16x32_bf16(false, A.v, false, Bq[1].v, (short)0, af, false, false);
    ao = __builtin_amdgcn_wmma_f32_16x16x32_bf16(false, A.v, false, Bq[2].v, (short)0, ao, false, false);
    ag = __builtin_amdgcn_wmma_f32_16x16x32_bf16(false, A.v, false, Bq[3].v, (short)0, ag, false, false);
  }

  // cell epilogue entirely in registers: C layout VGPR j -> row mBase+j+half*8, col nBase+r
  #pragma unroll
  for (int j = 0; j < 8; ++j) {
    const int row = mBase + j + half * 8;
    const int off = row * H_ + col;
    const float iv = sigm(ai[j]);
    const float fv = sigm(af[j]);
    const float ov = sigm(ao[j]);
    const float gv = tanh_f(ag[j]);
    const float cn = fv * c[off] + iv * gv;
    c[off] = cn;
    hnext[off] = (__bf16)(ov * tanh_f(cn));
  }
}

/* ------------- layernorm: one wave per batch row ------------- */
__device__ inline void layernorm_row(int row, int lane,
                                     const __bf16* __restrict__ h,
                                     __bf16* __restrict__ a,
                                     const float* __restrict__ g,
                                     const float* __restrict__ beta)
{
  float vals[16]; float s = 0.f, q = 0.f;
  #pragma unroll
  for (int m = 0; m < 16; ++m) {
    float v = (float)h[row * H_ + lane + m * 32];
    vals[m] = v; s += v; q += v * v;
  }
  #pragma unroll
  for (int m = 16; m > 0; m >>= 1) {
    s += __shfl_xor(s, m, 32);
    q += __shfl_xor(q, m, 32);
  }
  const float mu  = s * (1.0f / H_);
  const float var = q * (1.0f / H_) - mu * mu;
  const float rs  = rsqrtf(var + 1e-5f);
  #pragma unroll
  for (int m = 0; m < 16; ++m) {
    const int k = lane + m * 32;
    a[row * H_ + k] = (__bf16)((vals[m] - mu) * rs * g[k] + beta[k]);
  }
}

/* ------------- embedding gather for step t: one wave per batch row ------------- */
__device__ inline void embed_step(int brow, int lane, int t,
                                  const int* __restrict__ x,
                                  const __bf16* __restrict__ embbf,
                                  __bf16* __restrict__ xe)
{
  const int idx = x[brow * T_ + t];
  const uint4* src = (const uint4*)(embbf + (size_t)idx * H_);
  uint4* dst = (uint4*)(xe + (size_t)brow * H_);
  dst[lane * 2]     = src[lane * 2];
  dst[lane * 2 + 1] = src[lane * 2 + 1];
}

/* ------------- projection: waves 0..127, one 16x16 out tile each ------------- */
__device__ inline void proj_tile(int gw, int lane, int t,
                                 const __bf16* __restrict__ a,
                                 const __bf16* __restrict__ pWb,
                                 const float* __restrict__ pb,
                                 float* __restrict__ out)
{
  const int half = lane >> 4, r = lane & 15;
  const int mt = gw >> 4, nt = gw & 15;               // 8 x 16 tiles
  const int mBase = mt * 16, nBase = nt * 16;
  const int col = nBase + r;
  const float pv = pb[col];
  v8f acc = {pv, pv, pv, pv, pv, pv, pv, pv};

  #pragma unroll 4
  for (int kc = 0; kc < 16; ++kc) {
    const int k0 = (kc << 5) + half * 8;
    FragA A, Bq;
    { const uint4* p = (const uint4*)(a + (size_t)(mBase + r) * H_ + k0);
      A.q[0] = p[0]; A.q[1] = p[2]; }
    { const uint4* p = (const uint4*)(pWb + (size_t)(nBase + r) * H_ + k0);
      Bq.q[0] = p[0]; Bq.q[1] = p[2]; }
    acc = __builtin_amdgcn_wmma_f32_16x16x32_bf16(false, A.v, false, Bq.v, (short)0, acc, false, false);
  }
  #pragma unroll
  for (int j = 0; j < 8; ++j) {
    const int row = mBase + j + half * 8;
    out[(size_t)row * T_ * V_ + (size_t)t * V_ + col] = acc[j];
  }
}

/* ------------- prep: fp32 -> bf16 conversions + state zero-init ------------- */
__global__ void prep_kernel(const float* __restrict__ W, const float* __restrict__ pW,
                            const float* __restrict__ emb, char* __restrict__ ws)
{
  __bf16* Wbf   = (__bf16*)(ws + OFF_WBF);
  __bf16* pWbf  = (__bf16*)(ws + OFF_PWBF);
  __bf16* embbf = (__bf16*)(ws + OFF_EMBBF);
  __bf16* h0a   = (__bf16*)(ws + OFF_H0A);
  __bf16* h1a   = (__bf16*)(ws + OFF_H1A);
  float*  c0    = (float*)(ws + OFF_C0);
  float*  c1    = (float*)(ws + OFF_C1);
  unsigned* bar = (unsigned*)(ws + OFF_BAR);

  const size_t nW = (size_t)2 * NG * KZ;
  const size_t nP = (size_t)V_ * H_;
  const size_t nE = (size_t)256 * 512;
  const size_t nZ = (size_t)B_ * H_;
  const size_t total = nW + nP + nE + 4 * nZ + 64;

  size_t i = (size_t)blockIdx.x * blockDim.x + threadIdx.x;
  const size_t stride = (size_t)gridDim.x * blockDim.x;
  for (; i < total; i += stride) {
    size_t j = i;
    if (j < nW) { Wbf[j]   = (__bf16)W[j];   continue; } j -= nW;
    if (j < nP) { pWbf[j]  = (__bf16)pW[j];  continue; } j -= nP;
    if (j < nE) { embbf[j] = (__bf16)emb[j]; continue; } j -= nE;
    if (j < nZ) { h0a[j] = (__bf16)0.0f;     continue; } j -= nZ;
    if (j < nZ) { h1a[j] = (__bf16)0.0f;     continue; } j -= nZ;
    if (j < nZ) { c0[j] = 0.0f;              continue; } j -= nZ;
    if (j < nZ) { c1[j] = 0.0f;              continue; } j -= nZ;
    bar[j] = 0u;
  }
}

/* ------------- persistent recurrent kernel ------------- */
__global__ __launch_bounds__(NTHR, 1) void lstm_persist(
    const int* __restrict__ x, const float* __restrict__ bias,
    const float* __restrict__ ln_g, const float* __restrict__ ln_b,
    const float* __restrict__ pb, float* __restrict__ out, char* __restrict__ ws)
{
  const __bf16* Wbf   = (const __bf16*)(ws + OFF_WBF);
  const __bf16* pWbf  = (const __bf16*)(ws + OFF_PWBF);
  const __bf16* embbf = (const __bf16*)(ws + OFF_EMBBF);
  __bf16* xe  = (__bf16*)(ws + OFF_XE);
  __bf16* h0a = (__bf16*)(ws + OFF_H0A);
  __bf16* h0b = (__bf16*)(ws + OFF_H0B);
  __bf16* h1a = (__bf16*)(ws + OFF_H1A);
  __bf16* h1b = (__bf16*)(ws + OFF_H1B);
  __bf16* lna = (__bf16*)(ws + OFF_LNA);
  float*  c0  = (float*)(ws + OFF_C0);
  float*  c1  = (float*)(ws + OFF_C1);

  const int lane = threadIdx.x & 31;
  const int gw   = blockIdx.x * (NTHR / 32) + (threadIdx.x >> 5);   // 0..255

  if (gw >= 128) embed_step(gw - 128, lane, 0, x, embbf, xe);
  gbar(ws);

  #pragma unroll 1
  for (int t = 0; t < T_; ++t) {
    const bool even = (t & 1) == 0;
    __bf16* h0p = even ? h0a : h0b;
    __bf16* h0n = even ? h0b : h0a;
    __bf16* h1p = even ? h1a : h1b;
    __bf16* h1n = even ? h1b : h1a;

    // layer 0: z = [xe, h0_prev]
    gemm_cell(gw, lane, xe, h0p, h0n, c0, Wbf, bias);
    gbar(ws);
    // layer 1: z = [h0_new, h1_prev]
    gemm_cell(gw, lane, h0n, h1p, h1n, c1, Wbf + (size_t)NG * KZ, bias + NG);
    gbar(ws);
    // LN on waves 0..127; waves 128..255 concurrently gather embedding for t+1
    if (gw < 128)           layernorm_row(gw, lane, h1n, lna, ln_g, ln_b);
    else if (t + 1 < T_)    embed_step(gw - 128, lane, t + 1, x, embbf, xe);
    gbar(ws);
    // projection to vocab logits
    if (gw < 128) proj_tile(gw, lane, t, lna, pWbf, pb, out);
    gbar(ws);
  }
}

extern "C" void kernel_launch(void* const* d_in, const int* in_sizes, int n_in,
                              void* d_out, int out_size, void* d_ws, size_t ws_size,
                              hipStream_t stream) {
  const int*   x    = (const int*)d_in[0];
  const float* emb  = (const float*)d_in[1];
  const float* W    = (const float*)d_in[2];
  const float* bias = (const float*)d_in[3];
  const float* ln_g = (const float*)d_in[4];
  const float* ln_b = (const float*)d_in[5];
  const float* pW   = (const float*)d_in[6];
  const float* pb   = (const float*)d_in[7];
  float* out = (float*)d_out;
  char*  ws  = (char*)d_ws;

  prep_kernel<<<1024, 256, 0, stream>>>(W, pW, emb, ws);
  lstm_persist<<<NBLK, NTHR, 0, stream>>>(x, bias, ln_g, ln_b, pb, out, ws);
}
